// GlobalTopKAGP_44890998178035
// MI455X (gfx1250) — matrix operand order repaired and checked by reading, not compiled, not run on hardware
//
#include <hip/hip_runtime.h>
#include <math.h>

typedef __attribute__((ext_vector_type(16))) __bf16 v16bf;
typedef __attribute__((ext_vector_type(8)))  __bf16 v8bf;
typedef __attribute__((ext_vector_type(8)))  float  v8f;

#define BROWS   8192
#define DDIM    512
#define KTOP    8
#define TAU_INV 5.0f
#define ALPHA   0.3f
#define LN_EPS  1e-5f
#define NEG_INF (-3.4e38f)

// ---------------------------------------------------------------------------
// float -> bf16 (RNE) via bit manipulation; bf16 stored as unsigned short.
// ---------------------------------------------------------------------------
__device__ inline unsigned short f32_to_bf16_bits(float x) {
    unsigned u = __float_as_uint(x);
    u += 0x7FFFu + ((u >> 16) & 1u);
    return (unsigned short)(u >> 16);
}
__device__ inline float bf16_bits_to_f32(unsigned short h) {
    return __uint_as_float(((unsigned)h) << 16);
}

// ---------------------------------------------------------------------------
// Kernel 1: split fp32 rows into bf16 (hi, lo) pairs; optional L2 row-norm.
// blockDim = 256, D = 512 -> 2 elements/thread. gridDim.x = #rows.
// ---------------------------------------------------------------------------
__global__ void split_rows_kernel(const float* __restrict__ X,
                                  unsigned short* __restrict__ Hi,
                                  unsigned short* __restrict__ Lo,
                                  int do_norm) {
    __shared__ float red[8];
    const int row  = blockIdx.x;
    const int t    = threadIdx.x;
    const int lane = t & 31, wid = t >> 5;
    const float x0 = X[(size_t)row * DDIM + t];
    const float x1 = X[(size_t)row * DDIM + t + 256];
    float inv = 1.0f;
    if (do_norm) {
        float s = x0 * x0 + x1 * x1;
        #pragma unroll
        for (int off = 16; off > 0; off >>= 1) s += __shfl_xor(s, off, 32);
        if (lane == 0) red[wid] = s;
        __syncthreads();
        float tot = 0.f;
        #pragma unroll
        for (int i = 0; i < 8; ++i) tot += red[i];
        inv = 1.0f / fmaxf(sqrtf(tot), 1e-12f);
    }
    const float y0 = x0 * inv, y1 = x1 * inv;
    const unsigned short h0 = f32_to_bf16_bits(y0);
    const unsigned short h1 = f32_to_bf16_bits(y1);
    Hi[(size_t)row * DDIM + t]       = h0;
    Hi[(size_t)row * DDIM + t + 256] = h1;
    Lo[(size_t)row * DDIM + t]       = f32_to_bf16_bits(y0 - bf16_bits_to_f32(h0));
    Lo[(size_t)row * DDIM + t + 256] = f32_to_bf16_bits(y1 - bf16_bits_to_f32(h1));
}

// ---------------------------------------------------------------------------
// WMMA fragment loaders (CDNA5 16x16x32 bf16 layouts, wave32).
// A 16x32: lane l holds row (l&15); K = {8*half..8*half+7} U {16+8*half..+7}.
// B 32x16 (from row-major B[N,K], computing A @ B^T): lane l holds col (l&15),
//   K = {8*half..8*half+7} U {16+8*half..+7} (same as A).
// ---------------------------------------------------------------------------
__device__ inline v16bf load_afrag(const unsigned short* __restrict__ A, int ld,
                                   int row, int k0, int khalf) {
    const v8bf* p0 = reinterpret_cast<const v8bf*>(A + (size_t)row * ld + k0 + 8 * khalf);
    const v8bf* p1 = reinterpret_cast<const v8bf*>(A + (size_t)row * ld + k0 + 16 + 8 * khalf);
    v8bf a0 = *p0, a1 = *p1;
    v16bf r;
    #pragma unroll
    for (int e = 0; e < 8; ++e) { r[e] = a0[e]; r[e + 8] = a1[e]; }
    return r;
}
__device__ inline v16bf load_bfrag(const unsigned short* __restrict__ Bm, int ld,
                                   int col, int k0, int khalf) {
    const v8bf* p0 = reinterpret_cast<const v8bf*>(Bm + (size_t)col * ld + k0 + 8 * khalf);
    const v8bf* p1 = reinterpret_cast<const v8bf*>(Bm + (size_t)col * ld + k0 + 16 + 8 * khalf);
    v8bf b0 = *p0, b1 = *p1;
    v16bf r;
    #pragma unroll
    for (int e = 0; e < 8; ++e) { r[e] = b0[e]; r[e + 8] = b1[e]; }
    return r;
}

#define WMMA_BF16(A, B, C) \
    __builtin_amdgcn_wmma_f32_16x16x32_bf16(false, (A), false, (B), (short)0, (C), false, false)

// C/D f32 16x16: VGPR r, lane l -> M = r + 8*(l>>4), N = (l&15).
__device__ inline void store_tile(v8f c, float scale,
                                  float* __restrict__ C, int ldc,
                                  int row0, int col) {
    #pragma unroll
    for (int r = 0; r < 8; ++r) c[r] *= scale;
    #pragma unroll
    for (int r = 0; r < 8; ++r) C[(size_t)(row0 + r) * ldc + col] = c[r];
}

// ---------------------------------------------------------------------------
// Kernel 2: C[M,N] = scale * (A[M,K] @ B[N,K]^T), bf16x3 split precision.
// Each wave computes a 32x32 tile (2x2 of 16x16 WMMA tiles, 12 wmma / k-step).
// blockDim = 256 (8 waves).
// ---------------------------------------------------------------------------
__global__ void gemm_bf16x3_kernel(const unsigned short* __restrict__ Ah,
                                   const unsigned short* __restrict__ Al,
                                   const unsigned short* __restrict__ Bh,
                                   const unsigned short* __restrict__ Bl,
                                   float* __restrict__ C,
                                   int M, int N, int K, float scale) {
    const int lane  = threadIdx.x & 31;
    const int wave  = threadIdx.x >> 5;
    const int gw    = blockIdx.x * (blockDim.x >> 5) + wave;
    const int ntile = N >> 5;
    const int tm = gw / ntile, tn = gw % ntile;
    if (tm >= (M >> 5)) return;               // whole-wave uniform exit, EXEC stays full
    const int m0 = tm << 5, n0 = tn << 5;
    const int lrow  = lane & 15;
    const int khalf = lane >> 4;

    v8f c00 = {}, c01 = {}, c10 = {}, c11 = {};
    for (int k0 = 0; k0 < K; k0 += 32) {
        v16bf a0h = load_afrag(Ah, K, m0 + lrow,      k0, khalf);
        v16bf a1h = load_afrag(Ah, K, m0 + 16 + lrow, k0, khalf);
        v16bf a0l = load_afrag(Al, K, m0 + lrow,      k0, khalf);
        v16bf a1l = load_afrag(Al, K, m0 + 16 + lrow, k0, khalf);
        v16bf b0h = load_bfrag(Bh, K, n0 + lrow,      k0, khalf);
        v16bf b1h = load_bfrag(Bh, K, n0 + 16 + lrow, k0, khalf);
        v16bf b0l = load_bfrag(Bl, K, n0 + lrow,      k0, khalf);
        v16bf b1l = load_bfrag(Bl, K, n0 + 16 + lrow, k0, khalf);

        // x*y ~= xh*yh + xh*yl + xl*yh  (drop xl*yl)
        c00 = WMMA_BF16(a0h, b0h, c00);
        c00 = WMMA_BF16(a0h, b0l, c00);
        c00 = WMMA_BF16(a0l, b0h, c00);
        c01 = WMMA_BF16(a0h, b1h, c01);
        c01 = WMMA_BF16(a0h, b1l, c01);
        c01 = WMMA_BF16(a0l, b1h, c01);
        c10 = WMMA_BF16(a1h, b0h, c10);
        c10 = WMMA_BF16(a1h, b0l, c10);
        c10 = WMMA_BF16(a1l, b0h, c10);
        c11 = WMMA_BF16(a1h, b1h, c11);
        c11 = WMMA_BF16(a1h, b1l, c11);
        c11 = WMMA_BF16(a1l, b1h, c11);
    }
    const int r0 = m0 + 8 * khalf;
    store_tile(c00, scale, C, N, r0,      n0 + lrow);
    store_tile(c01, scale, C, N, r0,      n0 + 16 + lrow);
    store_tile(c10, scale, C, N, r0 + 16, n0 + lrow);
    store_tile(c11, scale, C, N, r0 + 16, n0 + 16 + lrow);
}

// ---------------------------------------------------------------------------
// Insert (v, j) into a per-thread descending sorted top-8 list.
// ---------------------------------------------------------------------------
__device__ inline void topk_insert(float v, int j, float (&tv)[KTOP], int (&ti)[KTOP]) {
    if (v > tv[KTOP - 1]) {
        tv[KTOP - 1] = v; ti[KTOP - 1] = j;
        #pragma unroll
        for (int t = KTOP - 1; t > 0; --t) {
            if (tv[t] > tv[t - 1]) {
                float fv = tv[t]; tv[t] = tv[t - 1]; tv[t - 1] = fv;
                int   fi = ti[t]; ti[t] = ti[t - 1]; ti[t - 1] = fi;
            }
        }
    }
}

// ---------------------------------------------------------------------------
// Kernel 3a: ROW top-8 + softmax. One wave per row of S (coalesced row scan).
// Writes dir-0 slot of Idx/W.
// ---------------------------------------------------------------------------
__global__ void row_topk_softmax_kernel(const float* __restrict__ S,
                                        int* __restrict__ IdxOut,
                                        float* __restrict__ WOut) {
    const int lane = threadIdx.x & 31;
    const int wave = threadIdx.x >> 5;
    const int row  = blockIdx.x * 8 + wave;
    const float* rp = S + (size_t)row * BROWS;

    float tv[KTOP]; int ti[KTOP];
    #pragma unroll
    for (int t = 0; t < KTOP; ++t) { tv[t] = NEG_INF; ti[t] = 0; }
    for (int j = lane; j < BROWS; j += 32) topk_insert(rp[j], j, tv, ti);

    // 8 rounds of wave-argmax; winner pops its list head (static indexing only)
    float selv[KTOP]; int seli[KTOP];
    #pragma unroll
    for (int t = 0; t < KTOP; ++t) {
        float bv = tv[0]; int bl = lane;
        #pragma unroll
        for (int off = 16; off > 0; off >>= 1) {
            float ov = __shfl_xor(bv, off, 32);
            int   ol = __shfl_xor(bl, off, 32);
            if (ov > bv || (ov == bv && ol < bl)) { bv = ov; bl = ol; }
        }
        const int widx = __shfl(ti[0], bl, 32);
        selv[t] = bv; seli[t] = widx;
        if (lane == bl) {
            #pragma unroll
            for (int u = 0; u < KTOP - 1; ++u) { tv[u] = tv[u + 1]; ti[u] = ti[u + 1]; }
            tv[KTOP - 1] = NEG_INF;
        }
    }
    const float m = selv[0];
    float w[KTOP], sum = 0.f;
    #pragma unroll
    for (int t = 0; t < KTOP; ++t) { w[t] = __expf(selv[t] - m); sum += w[t]; }
    const float inv = 1.0f / sum;
    if (lane == 0) {
        const size_t base = (size_t)row * KTOP;
        #pragma unroll
        for (int t = 0; t < KTOP; ++t) { IdxOut[base + t] = seli[t]; WOut[base + t] = w[t] * inv; }
    }
}

// ---------------------------------------------------------------------------
// Kernel 3b: COLUMN top-8 + softmax directly on S (no S^T materialization).
// Block owns 32 columns; thread t -> column (t&31), row-group (t>>5).
// Each wave reads one contiguous 128B row-slice per step (fully coalesced).
// 8 per-column sorted lists merged in LDS; softmax. Writes dir-1 slot.
// ---------------------------------------------------------------------------
__global__ void col_topk_softmax_kernel(const float* __restrict__ S,
                                        int* __restrict__ IdxOut,
                                        float* __restrict__ WOut) {
    __shared__ float sval[8][32][KTOP];
    __shared__ int   sidx[8][32][KTOP];
    const int t    = threadIdx.x;
    const int colg = t & 31;
    const int rg   = t >> 5;
    const int col  = blockIdx.x * 32 + colg;

    float tv[KTOP]; int ti[KTOP];
    #pragma unroll
    for (int k = 0; k < KTOP; ++k) { tv[k] = NEG_INF; ti[k] = 0; }
    for (int r = rg; r < BROWS; r += 8)
        topk_insert(S[(size_t)r * BROWS + col], r, tv, ti);

    #pragma unroll
    for (int k = 0; k < KTOP; ++k) { sval[rg][colg][k] = tv[k]; sidx[rg][colg][k] = ti[k]; }
    __syncthreads();

    if (t < 32) {   // one merge thread per column: 8-way sorted-list merge
        int p[8] = {0, 0, 0, 0, 0, 0, 0, 0};
        float selv[KTOP]; int seli[KTOP];
        #pragma unroll
        for (int k = 0; k < KTOP; ++k) {
            float bv = NEG_INF; int bg = 0;
            #pragma unroll
            for (int g = 0; g < 8; ++g) {
                const float v = sval[g][colg][p[g]];   // p[g] <= 7 at every read
                if (v > bv) { bv = v; bg = g; }
            }
            selv[k] = bv;
            seli[k] = sidx[bg][colg][p[bg]];
            p[bg]++;
        }
        const float m = selv[0];
        float w[KTOP], sum = 0.f;
        #pragma unroll
        for (int k = 0; k < KTOP; ++k) { w[k] = __expf(selv[k] - m); sum += w[k]; }
        const float inv = 1.0f / sum;
        const size_t base = ((size_t)BROWS + col) * KTOP;   // dir-1 slot
        #pragma unroll
        for (int k = 0; k < KTOP; ++k) { IdxOut[base + k] = seli[k]; WOut[base + k] = w[k] * inv; }
    }
}

// ---------------------------------------------------------------------------
// Kernel 4: fused sparse message gather + residual + LayerNorm. 1 block/row.
// ---------------------------------------------------------------------------
__global__ void msg_ln_kernel(const float* __restrict__ G, const float* __restrict__ P,
                              const int* __restrict__ Idx, const float* __restrict__ W,
                              const float* __restrict__ gamma, const float* __restrict__ beta,
                              float* __restrict__ Out) {
    __shared__ float rs[8], rq[8];
    const int row = blockIdx.x;
    const int t   = threadIdx.x;
    const int lane = t & 31, wid = t >> 5;

    int id[KTOP]; float wt[KTOP];
    #pragma unroll
    for (int j = 0; j < KTOP; ++j) {
        id[j] = Idx[(size_t)row * KTOP + j];
        wt[j] = W[(size_t)row * KTOP + j];
    }
    float m0 = 0.f, m1 = 0.f;
    #pragma unroll
    for (int j = 0; j < KTOP; ++j) {
        m0 += wt[j] * P[(size_t)id[j] * DDIM + t];
        m1 += wt[j] * P[(size_t)id[j] * DDIM + t + 256];
    }
    const float x0 = G[(size_t)row * DDIM + t]       + ALPHA * m0;
    const float x1 = G[(size_t)row * DDIM + t + 256] + ALPHA * m1;

    float s = x0 + x1, q = x0 * x0 + x1 * x1;
    #pragma unroll
    for (int off = 16; off > 0; off >>= 1) {
        s += __shfl_xor(s, off, 32);
        q += __shfl_xor(q, off, 32);
    }
    if (lane == 0) { rs[wid] = s; rq[wid] = q; }
    __syncthreads();
    float ts = 0.f, tq = 0.f;
    #pragma unroll
    for (int i = 0; i < 8; ++i) { ts += rs[i]; tq += rq[i]; }
    const float mu   = ts * (1.0f / DDIM);
    const float var  = tq * (1.0f / DDIM) - mu * mu;
    const float rstd = rsqrtf(var + LN_EPS);
    Out[(size_t)row * DDIM + t]       = (x0 - mu) * rstd * gamma[t]       + beta[t];
    Out[(size_t)row * DDIM + t + 256] = (x1 - mu) * rstd * gamma[t + 256] + beta[t + 256];
}

// ---------------------------------------------------------------------------
extern "C" void kernel_launch(void* const* d_in, const int* in_sizes, int n_in,
                              void* d_out, int out_size, void* d_ws, size_t ws_size,
                              hipStream_t stream) {
    (void)in_sizes; (void)n_in; (void)out_size; (void)ws_size;
    const float* gI      = (const float*)d_in[0];
    const float* gT      = (const float*)d_in[1];
    const float* Wi      = (const float*)d_in[2];
    const float* Wt      = (const float*)d_in[3];
    const float* gamma_i = (const float*)d_in[4];
    const float* beta_i  = (const float*)d_in[5];
    const float* gamma_t = (const float*)d_in[6];
    const float* beta_t  = (const float*)d_in[7];

    float* out      = (float*)d_out;
    float* out_gI2  = out;
    float* out_gT2  = out + (size_t)BROWS * DDIM;
    float* out_S    = out + 2 * (size_t)BROWS * DDIM;

    char* ws = (char*)d_ws;
    size_t cursor = 0;
    auto alloc = [&](size_t bytes) -> void* {
        void* p = ws + cursor;
        cursor += (bytes + 255) & ~(size_t)255;
        return p;
    };
    const size_t gbytes = (size_t)BROWS * DDIM * sizeof(unsigned short);
    const size_t wbytes = (size_t)DDIM * DDIM * sizeof(unsigned short);

    unsigned short* gi_h  = (unsigned short*)alloc(gbytes);
    unsigned short* gi_l  = (unsigned short*)alloc(gbytes);
    unsigned short* gt_h  = (unsigned short*)alloc(gbytes);
    unsigned short* gt_l  = (unsigned short*)alloc(gbytes);
    unsigned short* gIr_h = (unsigned short*)alloc(gbytes);
    unsigned short* gIr_l = (unsigned short*)alloc(gbytes);
    unsigned short* gTr_h = (unsigned short*)alloc(gbytes);
    unsigned short* gTr_l = (unsigned short*)alloc(gbytes);
    unsigned short* Wi_h  = (unsigned short*)alloc(wbytes);
    unsigned short* Wi_l  = (unsigned short*)alloc(wbytes);
    unsigned short* Wt_h  = (unsigned short*)alloc(wbytes);
    unsigned short* Wt_l  = (unsigned short*)alloc(wbytes);
    float* Pi   = (float*)alloc((size_t)BROWS * DDIM * sizeof(float));
    float* Pt   = (float*)alloc((size_t)BROWS * DDIM * sizeof(float));
    int*   idxb = (int*)alloc(2 * (size_t)BROWS * KTOP * sizeof(int));
    float* wb   = (float*)alloc(2 * (size_t)BROWS * KTOP * sizeof(float));

    // 1) bf16 hi/lo splits (normalized for S operands, raw for projections)
    split_rows_kernel<<<BROWS, 256, 0, stream>>>(gI, gi_h,  gi_l,  1);
    split_rows_kernel<<<BROWS, 256, 0, stream>>>(gT, gt_h,  gt_l,  1);
    split_rows_kernel<<<BROWS, 256, 0, stream>>>(gI, gIr_h, gIr_l, 0);
    split_rows_kernel<<<BROWS, 256, 0, stream>>>(gT, gTr_h, gTr_l, 0);
    split_rows_kernel<<<DDIM,  256, 0, stream>>>(Wi, Wi_h,  Wi_l,  0);
    split_rows_kernel<<<DDIM,  256, 0, stream>>>(Wt, Wt_h,  Wt_l,  0);

    // 2) S = gi @ gt^T / tau  (written once; both top-k passes read S directly)
    {
        const int waves = (BROWS / 32) * (BROWS / 32);
        gemm_bf16x3_kernel<<<waves / 8, 256, 0, stream>>>(
            gi_h, gi_l, gt_h, gt_l, out_S, BROWS, BROWS, DDIM, TAU_INV);
    }
    // 3) P_t = gT @ Wt^T ; P_i = gI @ Wi^T
    {
        const int waves = (BROWS / 32) * (DDIM / 32);
        gemm_bf16x3_kernel<<<waves / 8, 256, 0, stream>>>(
            gTr_h, gTr_l, Wt_h, Wt_l, Pt, BROWS, DDIM, DDIM, 1.0f);
        gemm_bf16x3_kernel<<<waves / 8, 256, 0, stream>>>(
            gIr_h, gIr_l, Wi_h, Wi_l, Pi, BROWS, DDIM, DDIM, 1.0f);
    }
    // 4) top-8 + softmax: rows of S (dir 0), columns of S (dir 1, coalesced tiles)
    row_topk_softmax_kernel<<<BROWS / 8, 256, 0, stream>>>(out_S, idxb, wb);
    col_topk_softmax_kernel<<<BROWS / 32, 256, 0, stream>>>(out_S, idxb, wb);

    // 5) fused sparse gather + residual + LayerNorm
    msg_ln_kernel<<<BROWS, 256, 0, stream>>>(gI, Pt, idxb, wb,
                                             gamma_i, beta_i, out_gI2);
    msg_ln_kernel<<<BROWS, 256, 0, stream>>>(gT, Pi, idxb + (size_t)BROWS * KTOP,
                                             wb + (size_t)BROWS * KTOP,
                                             gamma_t, beta_t, out_gT2);
}
